// QuantizedViTOutput_48318382080289
// MI455X (gfx1250) — matrix operand order repaired and checked by reading, not compiled, not run on hardware
//
#include <hip/hip_runtime.h>

// ---------------------------------------------------------------------------
// QuantizedViTOutput for MI455X (gfx1250, wave32, WMMA + async-to-LDS)
//   out = (hidden @ w_fq^T) + bias + residual
//   w_fq = round(W/scale).clip(-128,127) * scale,  scale = max|W_row|/127
// w_int is an exact small integer -> bf16-exact. GEMM via
// V_WMMA_F32_16X16X32_BF16 with fp32 accumulation; per-channel scale applied
// in the fp32 epilogue. B tiles move HBM->LDS with GLOBAL_LOAD_ASYNC_TO_LDS
// (ASYNCcnt); A tiles convert fp32->bf16 through VGPRs. Double-buffered LDS,
// one barrier per K-step.
// ---------------------------------------------------------------------------

typedef __attribute__((ext_vector_type(16))) __bf16 v16bf;
typedef __attribute__((ext_vector_type(8)))  __bf16 v8bf;
typedef __attribute__((ext_vector_type(8)))  float  v8f;

#define K_DIM 4096   // D_FF
#define N_DIM 1024   // D
#define LDA   40     // padded LDS row stride (halves): 80B rows, 16B-aligned frags
#define LDB   40

// ---------------------------------------------------------------------------
// Kernel 1: per-output-channel int8 symmetric fake-quant of W -> bf16 ints +
// fp32 scales, written to workspace.
// ---------------------------------------------------------------------------
__global__ __launch_bounds__(256) void quant_prep_kernel(
    const float* __restrict__ W,      // [N_DIM, K_DIM]
    __bf16* __restrict__ wq,          // [N_DIM, K_DIM] (integer values in bf16)
    float* __restrict__ scales)       // [N_DIM]
{
    const int d = blockIdx.x;
    const float* row = W + (size_t)d * K_DIM;

    float m = 0.0f;
#pragma unroll
    for (int i = 0; i < 16; ++i)
        m = fmaxf(m, fabsf(row[threadIdx.x + i * 256]));

    __shared__ float red[256];
    red[threadIdx.x] = m;
    __syncthreads();
#pragma unroll
    for (int s = 128; s > 0; s >>= 1) {
        if (threadIdx.x < s)
            red[threadIdx.x] = fmaxf(red[threadIdx.x], red[threadIdx.x + s]);
        __syncthreads();
    }
    const float maxabs = fmaxf(red[0], 1e-8f);
    const float scale  = maxabs * (1.0f / 127.0f);
    const float inv    = 127.0f / maxabs;

#pragma unroll
    for (int i = 0; i < 16; ++i) {
        const int idx = threadIdx.x + i * 256;
        float q = rintf(row[idx] * inv);           // round-half-even, like jnp.round
        q = fminf(fmaxf(q, -128.0f), 127.0f);
        wq[(size_t)d * K_DIM + idx] = (__bf16)q;   // exact: |q| <= 128
    }
    if (threadIdx.x == 0) scales[d] = scale;
}

// ---------------------------------------------------------------------------
// Kernel 2: GEMM + fused epilogue.
//   Block tile 128(M) x 128(N); 8 wave32 arranged 2(M) x 4(N);
//   wave tile 64x32 = 4x2 WMMA 16x16 accumulators; K step 32.
// ---------------------------------------------------------------------------
__global__ __launch_bounds__(256) void ffn_wmma_kernel(
    const float*  __restrict__ A,      // hidden  [M, K_DIM] fp32
    const float*  __restrict__ R,      // residual[M, N_DIM] fp32
    const __bf16* __restrict__ Wq,     // [N_DIM, K_DIM] bf16 (integers)
    const float*  __restrict__ bias,   // [N_DIM]
    const float*  __restrict__ scales, // [N_DIM]
    float* __restrict__ out)           // [M, N_DIM] fp32
{
    __shared__ __bf16 sA[2][128 * LDA];
    __shared__ __bf16 sB[2][128 * LDB];

    const int tid  = threadIdx.x;
    const int wave = tid >> 5;
    const int lane = tid & 31;
    const int half = lane >> 4;   // 0: lanes 0-15, 1: lanes 16-31
    const int lm   = lane & 15;
    const int mw   = wave >> 2;   // 0..1  (64 rows each)
    const int nw   = wave & 3;    // 0..3  (32 cols each)

    const int tileN = blockIdx.x * 128;
    const int tileM = blockIdx.y * 128;

    v8f acc[4][2] = {};           // [mi][ni] 16x16 fp32 accumulators

    float4 aReg[4];               // staged global A (fp32), converted on store

    // B tile: 128 x 32 bf16 = 8KB, fetched async HBM/L2 -> LDS (ASYNCcnt).
    // Each thread issues 2 x b128 (16B/lane) transfers.
    auto asyncB = [&](int bufSel, int k0) {
#pragma unroll
        for (int i = 0; i < 2; ++i) {
            const int e   = tid + i * 256;     // 0..511 uint4 slots (128 x 4)
            const int row = e >> 2;
            const int c8  = e & 3;
            const unsigned voff =
                (unsigned)(((tileN + row) * K_DIM + k0 + c8 * 8) * 2); // bytes
            const unsigned lds =
                (unsigned)(uintptr_t)&sB[bufSel][row * LDB + c8 * 8];  // LDS byte addr
            asm volatile("global_load_async_to_lds_b128 %0, %1, %2"
                         :: "v"(lds), "v"(voff), "s"(Wq)
                         : "memory");
        }
    };

    auto loadA = [&](int k0) {
#pragma unroll
        for (int i = 0; i < 4; ++i) {
            const int e   = tid + i * 256;     // 0..1023 float4 slots (128 x 8)
            const int row = e >> 3;
            const int c4  = e & 7;
            aReg[i] = *(const float4*)(A + (size_t)(tileM + row) * K_DIM + k0 + c4 * 4);
        }
    };
    auto storeA = [&](int bufSel) {
#pragma unroll
        for (int i = 0; i < 4; ++i) {
            const int e   = tid + i * 256;
            const int row = e >> 3;
            const int c4  = e & 7;
            __bf16* p = &sA[bufSel][row * LDA + c4 * 4];
            p[0] = (__bf16)aReg[i].x;
            p[1] = (__bf16)aReg[i].y;
            p[2] = (__bf16)aReg[i].z;
            p[3] = (__bf16)aReg[i].w;
        }
    };

    // A fragment: 16x32 bf16. lane<16: M=lm, K = 0..7,16..23 ; lane>=16: K = 8..15,24..31
    auto fragA = [&](const __bf16* base, int mi) -> v16bf {
        const __bf16* p = base + (mw * 64 + mi * 16 + lm) * LDA + half * 8;
        v8bf lo = *(const v8bf*)(p);
        v8bf hi = *(const v8bf*)(p + 16);
        return __builtin_shufflevector(lo, hi, 0,1,2,3,4,5,6,7,8,9,10,11,12,13,14,15);
    };
    // B fragment: 32x16 bf16. lane<16: N=lm, K=0..7,16..23 ; lane>=16: N=lm, K=8..15,24..31
    auto fragB = [&](const __bf16* base, int ni) -> v16bf {
        const __bf16* p = base + (nw * 32 + ni * 16 + lm) * LDB + half * 8;
        v8bf lo = *(const v8bf*)(p);
        v8bf hi = *(const v8bf*)(p + 16);
        return __builtin_shufflevector(lo, hi, 0,1,2,3,4,5,6,7,8,9,10,11,12,13,14,15);
    };

    // prologue: stage K-tile 0 into buffer 0
    asyncB(0, 0);
    loadA(0);
    storeA(0);
    asm volatile("s_wait_asynccnt 0x0" ::: "memory");
    __syncthreads();

    int buf = 0;
    const int KT = K_DIM / 32;   // 128 iterations
    for (int kt = 0; kt < KT; ++kt) {
        const bool more = (kt + 1) < KT;
        if (more) {                        // prefetch K-tile kt+1 into buf^1
            asyncB(buf ^ 1, (kt + 1) * 32);
            loadA((kt + 1) * 32);
        }

        const __bf16* cA = &sA[buf][0];
        const __bf16* cB = &sB[buf][0];
        v16bf bf0 = fragB(cB, 0);
        v16bf bf1 = fragB(cB, 1);
#pragma unroll
        for (int mi = 0; mi < 4; ++mi) {
            v16bf af = fragA(cA, mi);
            acc[mi][0] = __builtin_amdgcn_wmma_f32_16x16x32_bf16(
                false, af, false, bf0, (short)0, acc[mi][0], false, false);
            acc[mi][1] = __builtin_amdgcn_wmma_f32_16x16x32_bf16(
                false, af, false, bf1, (short)0, acc[mi][1], false, false);
        }

        if (more) storeA(buf ^ 1);         // other buffer: no extra barrier needed

        asm volatile("s_wait_asynccnt 0x0" ::: "memory");  // own async B landed
        __syncthreads();                   // everyone's LDS writes visible
        buf ^= 1;
    }

    // Epilogue: out = acc*scale[n] + bias[n] + residual
#pragma unroll
    for (int mi = 0; mi < 4; ++mi) {
#pragma unroll
        for (int ni = 0; ni < 2; ++ni) {
            const int n  = tileN + nw * 32 + ni * 16 + lm;
            const float sc = scales[n];
            const float bs = bias[n];
            const int mBase = tileM + mw * 64 + mi * 16 + half * 8;
#pragma unroll
            for (int v = 0; v < 8; ++v) {
                const size_t idx = (size_t)(mBase + v) * N_DIM + n;
                out[idx] = acc[mi][ni][v] * sc + bs + R[idx];
            }
        }
    }
}

// ---------------------------------------------------------------------------
extern "C" void kernel_launch(void* const* d_in, const int* in_sizes, int n_in,
                              void* d_out, int out_size, void* d_ws, size_t ws_size,
                              hipStream_t stream) {
    const float* hidden = (const float*)d_in[0];   // [16,1024,4096]
    const float* resid  = (const float*)d_in[1];   // [16,1024,1024]
    const float* W      = (const float*)d_in[2];   // [1024,4096]
    const float* bias   = (const float*)d_in[3];   // [1024]

    const int M = in_sizes[0] / K_DIM;             // 16384

    __bf16* wq     = (__bf16*)d_ws;                                        // 8 MB
    float*  scales = (float*)((char*)d_ws + (size_t)N_DIM * K_DIM * 2);    // 4 KB

    quant_prep_kernel<<<N_DIM, 256, 0, stream>>>(W, wq, scales);

    dim3 grid(N_DIM / 128, M / 128);               // (8, 128): x-fastest -> A panel L2 reuse
    ffn_wmma_kernel<<<grid, 256, 0, stream>>>(hidden, resid, wq, bias, scales,
                                              (float*)d_out);
}
